// PatchdistributionsLoss_12154757448088
// MI455X (gfx1250) — matrix operand order repaired and checked by reading, not compiled, not run on hardware
//
#include <hip/hip_runtime.h>
#include <hip/hip_bf16.h>

typedef __attribute__((ext_vector_type(16))) _Float16 v16h;
typedef __attribute__((ext_vector_type(8)))  float    v8f;

#define B_       16
#define H_       128
#define W_       128
#define PATCH_   7
#define GRID_    122           // (128-7)/1+1
#define L_       (GRID_*GRID_) // 14884
#define PDIM_    147
#define KSTEPS_  5             // 147 padded to 5 * 32
#define NSAMP_   4096
#define NPROJ_   128
#define SWZ_     (KSTEPS_*32*16)  // 2560 f16 per swizzled operand tile
#define ENTRIES_ (16.0f*14884.0f*147.0f)  // 35007168

__device__ __forceinline__ int imin(int a, int b) { return a < b ? a : b; }
__device__ __forceinline__ int imax(int a, int b) { return a > b ? a : b; }

// ---------------------------------------------------------------------------
// ws float layout: [0]=sum_x  [1]=sum_y  [2]=loss accumulator
// PX at ws+64 floats (32MB), PY after PX (32MB)
// ---------------------------------------------------------------------------

__global__ void zero_ws_kernel(float* wsf) {
    if (threadIdx.x == 0) { wsf[0] = 0.0f; wsf[1] = 0.0f; wsf[2] = 0.0f; }
}

// Coverage-weighted pixel sums == sum over all patch entries (patches overlap).
__global__ void mean_kernel(const float* __restrict__ x, const float* __restrict__ y,
                            float* __restrict__ wsf) {
    __shared__ float rx[256], ry[256];
    float sx = 0.0f, sy = 0.0f;
    const int N = B_ * 3 * H_ * W_;
    for (int i = blockIdx.x * blockDim.x + threadIdx.x; i < N; i += gridDim.x * blockDim.x) {
        int w = i & (W_ - 1);
        int h = (i >> 7) & (H_ - 1);
        float cw = (float)(imin(w, GRID_ - 1) - imax(w - (PATCH_ - 1), 0) + 1);
        float ch = (float)(imin(h, GRID_ - 1) - imax(h - (PATCH_ - 1), 0) + 1);
        float wt = cw * ch;
        sx += x[i] * wt;
        sy += y[i] * wt;
    }
    rx[threadIdx.x] = sx; ry[threadIdx.x] = sy;
    __syncthreads();
    for (int s = 128; s > 0; s >>= 1) {
        if ((int)threadIdx.x < s) {
            rx[threadIdx.x] += rx[threadIdx.x + s];
            ry[threadIdx.x] += ry[threadIdx.x + s];
        }
        __syncthreads();
    }
    if (threadIdx.x == 0) { atomicAdd(&wsf[0], rx[0]); atomicAdd(&wsf[1], ry[0]); }
}

// Gather sampled 7x7x3 patches, subtract global mean, project onto 128 unit
// directions with v_wmma_f32_16x16x32_f16. Operand tiles are staged in LDS in
// the exact per-lane WMMA register image, so each k-step operand fetch is one
// 32-byte LDS load (2x ds_load_b128) straight into the WMMA source VGPRs.
__global__ void project_kernel(const float* __restrict__ x, const float* __restrict__ y,
                               const float* __restrict__ proj,
                               const int* __restrict__ idx_x, const int* __restrict__ idx_y,
                               const float* __restrict__ wsf,
                               float* __restrict__ PX, float* __restrict__ PY) {
    __shared__ _Float16 Aswz[8][SWZ_];  // per-wave A tile in register image (5KB each)
    __shared__ _Float16 Bswz[SWZ_];     // shared B tile in register image (5KB)

    int g = blockIdx.x;
    int ngrp = g & 31; g >>= 5;   // 32 sample-tile groups (8 tiles each)
    int pt   = g & 7;  g >>= 3;   // 8 proj tiles of 16 columns
    int b    = g & 15; g >>= 4;   // batch
    int sel  = g;                 // 0 = x, 1 = y

    const float* src = sel ? y : x;
    const int*   idx = sel ? idx_y : idx_x;
    float*      outp = sel ? PY : PX;
    const float meanv = wsf[sel] * (1.0f / ENTRIES_);

    const int tid  = threadIdx.x;
    const int wid  = tid >> 5;    // wave32
    const int lane = tid & 31;

    // B fill: slot (kk, lt, ele) holds proj[K][pt*16+nn] with
    //   nn = lt&15, K = kk*32 + ((ele&8)<<1) + (lt>>4)*8 + (ele&7)   (matches A k order)
    for (int e = tid; e < SWZ_; e += 256) {
        int kk  = e >> 9;
        int rem = e & 511;
        int lt  = rem >> 4;
        int ele = rem & 15;
        int K   = kk * 32 + ((ele & 8) << 1) + ((lt >> 4) << 3) + (ele & 7);
        float v = (K < PDIM_) ? proj[K * NPROJ_ + pt * 16 + (lt & 15)] : 0.0f;
        Bswz[e] = (_Float16)v;
    }

    // A fill (per wave): slot (kk, lt, ele) holds patch element (m, K) with
    //   m = lt&15, kloc = ((ele&8)<<1) + (lt>>4)*8 + (ele&7)  (ISA 16-bit A 16x32)
    const int ntile = ngrp * 8 + wid;
    for (int e = lane; e < SWZ_; e += 32) {
        int kk   = e >> 9;
        int rem  = e & 511;
        int lt   = rem >> 4;
        int ele  = rem & 15;
        int m    = lt & 15;
        int kloc = ((ele & 8) << 1) + ((lt >> 4) << 3) + (ele & 7);
        int K    = kk * 32 + kloc;
        float v  = 0.0f;
        if (K < PDIM_) {
            int l  = idx[ntile * 16 + m];
            int h0 = l / GRID_;
            int w0 = l - h0 * GRID_;
            int c  = K / 49;
            int dd = K - c * 49;
            int rr = dd / 7;
            int cc = dd - rr * 7;
            v = src[(((b * 3 + c) * H_) + (h0 + rr)) * W_ + (w0 + cc)] - meanv;
        }
        Aswz[wid][e] = (_Float16)v;
    }
    __syncthreads();

    v8f acc = {};
    #pragma unroll
    for (int kk = 0; kk < KSTEPS_; ++kk) {
        v16h a  = *(const v16h*)&Aswz[wid][(kk * 32 + lane) * 16];
        v16h bv = *(const v16h*)&Bswz[(kk * 32 + lane) * 16];
        acc = __builtin_amdgcn_wmma_f32_16x16x32_f16(
            false, a, false, bv, (short)0, acc, false, false);
    }

    // C/D layout: VGPR r, lanes 0-15 -> M=r, lanes 16-31 -> M=r+8; N = lane%16.
    // All 8 values of a lane hit consecutive rows of ONE column -> 2x b128 store.
    const int nn = lane & 15;
    const int m0 = (lane >> 4) * 8;
    size_t off = ((size_t)(b * NPROJ_ + pt * 16 + nn)) * NSAMP_ + (size_t)(ntile * 16 + m0);
    float4 lo = make_float4(acc[0], acc[1], acc[2], acc[3]);
    float4 hi = make_float4(acc[4], acc[5], acc[6], acc[7]);
    *(float4*)(outp + off)     = lo;
    *(float4*)(outp + off + 4) = hi;
}

// One block per (b, p) column: bitonic-sort the 4096-sample x and y columns in
// LDS, then accumulate sum |sorted_x - sorted_y| into the global loss.
__global__ void sort_reduce_kernel(const float* __restrict__ PX, const float* __restrict__ PY,
                                   float* __restrict__ loss) {
    __shared__ float sx[NSAMP_];
    __shared__ float sy[NSAMP_];
    __shared__ float red[256];

    const int bi = blockIdx.x;
    const int p  = bi & (NPROJ_ - 1);
    const int b  = bi >> 7;
    const float* cx = PX + ((size_t)(b * NPROJ_ + p)) * NSAMP_;
    const float* cy = PY + ((size_t)(b * NPROJ_ + p)) * NSAMP_;

    for (int i = threadIdx.x; i < NSAMP_; i += 256) { sx[i] = cx[i]; sy[i] = cy[i]; }
    __syncthreads();

    for (int k = 2; k <= NSAMP_; k <<= 1) {
        for (int j = k >> 1; j > 0; j >>= 1) {
            for (int i = threadIdx.x; i < NSAMP_; i += 256) {
                int ixj = i ^ j;
                if (ixj > i) {
                    bool up = ((i & k) == 0);
                    float a0 = sx[i], a1 = sx[ixj];
                    if ((a0 > a1) == up) { sx[i] = a1; sx[ixj] = a0; }
                    float b0 = sy[i], b1 = sy[ixj];
                    if ((b0 > b1) == up) { sy[i] = b1; sy[ixj] = b0; }
                }
            }
            __syncthreads();
        }
    }

    float s = 0.0f;
    for (int i = threadIdx.x; i < NSAMP_; i += 256) s += fabsf(sx[i] - sy[i]);
    red[threadIdx.x] = s;
    __syncthreads();
    for (int st = 128; st > 0; st >>= 1) {
        if ((int)threadIdx.x < st) red[threadIdx.x] += red[threadIdx.x + st];
        __syncthreads();
    }
    if (threadIdx.x == 0) atomicAdd(loss, red[0]);
}

__global__ void final_kernel(const float* __restrict__ wsf, float* __restrict__ out) {
    if (threadIdx.x == 0) out[0] = wsf[2] * (1.0f / 8388608.0f);  // 16*4096*128 = 2^23
}

extern "C" void kernel_launch(void* const* d_in, const int* in_sizes, int n_in,
                              void* d_out, int out_size, void* d_ws, size_t ws_size,
                              hipStream_t stream) {
    const float* x     = (const float*)d_in[0];
    const float* y     = (const float*)d_in[1];
    const float* proj  = (const float*)d_in[2];
    const int*   idx_x = (const int*)d_in[3];
    const int*   idx_y = (const int*)d_in[4];
    float* out = (float*)d_out;

    float* wsf = (float*)d_ws;
    float* PX  = wsf + 64;                               // 256-byte aligned
    float* PY  = PX + (size_t)B_ * NPROJ_ * NSAMP_;      // +32 MB

    zero_ws_kernel<<<1, 32, 0, stream>>>(wsf);
    mean_kernel<<<1024, 256, 0, stream>>>(x, y, wsf);
    // 2 inputs * 16 batch * 8 proj-tiles * 32 sample-tile-groups = 8192 blocks
    project_kernel<<<8192, 256, 0, stream>>>(x, y, proj, idx_x, idx_y, wsf, PX, PY);
    sort_reduce_kernel<<<B_ * NPROJ_, 256, 0, stream>>>(PX, PY, &wsf[2]);
    final_kernel<<<1, 32, 0, stream>>>(wsf, out);
}
